// TensorLayer_43774306681422
// MI455X (gfx1250) — matrix operand order repaired and checked by reading, not compiled, not run on hardware
//
#include <hip/hip_runtime.h>

typedef _Float16 v16h __attribute__((ext_vector_type(16)));
typedef _Float16 v8h  __attribute__((ext_vector_type(8)));
typedef _Float16 v4h  __attribute__((ext_vector_type(4)));
typedef _Float16 v2h  __attribute__((ext_vector_type(2)));
typedef float    v8f  __attribute__((ext_vector_type(8)));

static __device__ __forceinline__ v2h cvt2h(float x, float y) {
    // __builtin_amdgcn_cvt_pkrtz returns __fp16 ext_vector(2); bit-cast to _Float16 vec.
    return __builtin_bit_cast(v2h, __builtin_amdgcn_cvt_pkrtz(x, y));
}
static __device__ __forceinline__ v8h splat8(_Float16 x) {
    return (v8h){x, x, x, x, x, x, x, x};
}
static __device__ __forceinline__ v4h splat4(_Float16 x) {
    return (v4h){x, x, x, x};
}

// data:   (8192, 32, 32, 4) f32
// weight: (16,16,4,4,4,4,8) f32  == W[m,n,p,q,u] with p=i*4+j, q=k*4+l
// out:    (8192, 16, 16, 8) f32, L2-normalized over u
//
// Per (sample s, tile t): out[u] = sum_{pq} (ab ⊗ cd)[pq] * W[t, pq, u]
// -> K=256 GEMM via 8 x V_WMMA_F32_16X16X32_F16 (f16 A/B, f32 accum).

__global__ __launch_bounds__(256) void tensor_layer_wmma(
    const float* __restrict__ data,
    const float* __restrict__ weight,
    float* __restrict__ out)
{
    const int tid  = threadIdx.x;
    const int lane = tid & 31;
    const int wv   = tid >> 5;        // wave in block: 0..7
    const int hi   = lane >> 4;       // lane half
    const int col  = lane & 15;       // B/D column (u for col<8)
    const int t    = blockIdx.x;      // tile 0..255
    const int m    = t >> 4;
    const int n    = t & 15;

    // ---------------- B matrices (weights), loaded once per block ----------------
    // B element e, lane L: K = 8*hi + (e&7) + 16*(e>>3), column = L&15.
    // Step ks: pq = 32*ks + K  ->  p = 2*ks + (e>>3), q = 8*hi + (e&7), u = col (cols >=8 zero pad).
    const size_t tileBase = (size_t)t * 2048;
    v16h Bv[8];
    #pragma unroll
    for (int ks = 0; ks < 8; ++ks) {
        const int p = 2 * ks;
        #pragma unroll
        for (int e = 0; e < 16; ++e) {
            float w = (col < 8)
                ? weight[tileBase + (size_t)(((p + (e >> 3)) * 16 + 8 * hi + (e & 7)) * 8 + col)]
                : 0.0f;
            Bv[ks][e] = (_Float16)w;
        }
    }

    const int srow = lane & 15;       // A row index within the 16-sample chunk

    #pragma unroll 1
    for (int iter = 0; iter < 8; ++iter) {
        const int s0 = ((iter * 8 + blockIdx.y) * 8 + wv) << 4;  // chunk base sample
        const int s  = s0 + srow;

        // a,b at (y=2m, x=2n..2n+1); c,d at (y=2m+1, x=2n..2n+1): 2x 32B contiguous.
        const float4* row0 = (const float4*)(data + (((size_t)s * 32 + 2 * m)     * 32 + 2 * n) * 4);
        const float4* row1 = (const float4*)(data + (((size_t)s * 32 + 2 * m + 1) * 32 + 2 * n) * 4);
        const float4 av = row0[0];
        const float4 bv = row0[1];
        const float4 cv = row1[0];
        const float4 dv = row1[1];

        if (iter < 7) {  // prefetch next chunk (s advances by 1024 samples)
            __builtin_prefetch((const float*)row0 + (size_t)1024 * 32 * 32 * 4, 0, 0);
            __builtin_prefetch((const float*)row1 + (size_t)1024 * 32 * 32 * 4, 0, 0);
        }

        // packed f32 -> f16 conversions (v_cvt_pkrtz_f16_f32)
        const v2h a01 = cvt2h(av.x, av.y);
        const v2h a23 = cvt2h(av.z, av.w);
        const v2h b01 = cvt2h(bv.x, bv.y);
        const v2h b23 = cvt2h(bv.z, bv.w);
        const v2h c01 = cvt2h(cv.x, cv.y);
        const v2h c23 = cvt2h(cv.z, cv.w);
        const v2h d01 = cvt2h(dv.x, dv.y);
        const v2h d23 = cvt2h(dv.z, dv.w);

        // cd values this lane half owns: q = 8*hi + k, k=0..7
        //   cdv = { ch[2*hi]*dh[0..3], ch[2*hi+1]*dh[0..3] }  (packed v4h muls)
        const v4h dh4 = __builtin_shufflevector(d01, d23, 0, 1, 2, 3);
        const _Float16 cA = hi ? c23[0] : c01[0];
        const _Float16 cB = hi ? c23[1] : c01[1];
        const v4h cdLo = dh4 * splat4(cA);
        const v4h cdHi = dh4 * splat4(cB);
        const v8h cdv  = __builtin_shufflevector(cdLo, cdHi, 0, 1, 2, 3, 4, 5, 6, 7);

        // ab outer product (16 scalars, constant-indexed only)
        const _Float16 ahs[4] = {a01[0], a01[1], a23[0], a23[1]};
        const _Float16 bhs[4] = {b01[0], b01[1], b23[0], b23[1]};
        _Float16 abh[16];
        #pragma unroll
        for (int i = 0; i < 4; ++i)
            #pragma unroll
            for (int j = 0; j < 4; ++j)
                abh[i * 4 + j] = ahs[i] * bhs[j];

        // ---------------- K=256 GEMM: 8 chained WMMA steps ----------------
        // A element e<8:  ab[2ks]   * cd[8*hi + e]
        //   element e>=8: ab[2ks+1] * cd[8*hi + e-8]   (packed v8h muls)
        v8f acc = {};
        #pragma unroll
        for (int ks = 0; ks < 8; ++ks) {
            const v8h lo = cdv * splat8(abh[2 * ks]);
            const v8h hp = cdv * splat8(abh[2 * ks + 1]);
            const v16h Av = __builtin_shufflevector(
                lo, hp, 0, 1, 2, 3, 4, 5, 6, 7, 8, 9, 10, 11, 12, 13, 14, 15);
            acc = __builtin_amdgcn_wmma_f32_16x16x32_f16(
                /*neg_a=*/false, Av, /*neg_b=*/false, Bv[ks],
                /*c_mod=*/(short)0, acc, /*reuse_a=*/false, /*reuse_b=*/false);
        }

        // ---------------- normalize over u (butterfly over lanes 0..7) ----------------
        // D element r, lane L: sample row = r + 8*hi, column u = col.
        float accN[8];
        #pragma unroll
        for (int r = 0; r < 8; ++r) {
            const float v = acc[r];
            float ss = v * v;
            ss += __shfl_xor(ss, 1, 32);
            ss += __shfl_xor(ss, 2, 32);
            ss += __shfl_xor(ss, 4, 32);
            const float sc = 1.0f / fmaxf(sqrtf(ss), 1e-12f);
            accN[r] = v * sc;
        }

        if (col < 8) {
            #pragma unroll
            for (int r = 0; r < 8; ++r) {
                const int sOut = s0 + r + 8 * hi;
                // out flat: ((s*16+m)*16+n)*8 + u = s*2048 + t*8 + u
                out[(size_t)sOut * 2048 + (size_t)t * 8 + col] = accN[r];
            }
        }
    }
}

extern "C" void kernel_launch(void* const* d_in, const int* in_sizes, int n_in,
                              void* d_out, int out_size, void* d_ws, size_t ws_size,
                              hipStream_t stream) {
    const float* data   = (const float*)d_in[0];
    const float* weight = (const float*)d_in[1];
    float*       out    = (float*)d_out;

    dim3 grid(256, 8, 1);   // 256 tiles x 8 sample groups; 8 waves/block x 8 iters
    dim3 block(256, 1, 1);
    tensor_layer_wmma<<<grid, block, 0, stream>>>(data, weight, out);
}
